// AllegroQeqLayer_54674933678514
// MI455X (gfx1250) — matrix operand order repaired and checked by reading, not compiled, not run on hardware
//
#include <hip/hip_runtime.h>

typedef __attribute__((ext_vector_type(16))) _Float16 v16h;
typedef __attribute__((ext_vector_type(8)))  _Float16 v8h;
typedef __attribute__((ext_vector_type(8)))  float    v8f;
typedef __attribute__((ext_vector_type(4)))  float    v4f;

#define F_IN 48
#define CE   16
#define H_DIM 64
#define HB_STRIDE 72   // 16B-padded row stride (halves) -> conflict-free b128 TR reads

// Fast silu: x * rcp(1+exp(-x)) using hardware v_rcp_f32 (no IEEE div sequence).
__device__ __forceinline__ float fast_silu(float h) {
    return h * __builtin_amdgcn_rcpf(1.f + __expf(-h));
}

// ---------------------------------------------------------------------------
// K0: convert W_x1..3 (fp32, pre-scaled by 1/sqrt(64)=0.125) into f16 WMMA
//     B-fragment layout; zero the chi accumulator and the pot output slot.
// frag flat index: (((layer*2+kb)*4+nb)*32 + lane)*16 + e
// value = W[layer][ (kb*32 + (lane>>4)*8 + (e&7) + (e>>3)*16) * 64 + (nb*16 + (lane&15)) ]
// ---------------------------------------------------------------------------
__global__ void __launch_bounds__(256) k0_prep(
    const float* __restrict__ Wx1, const float* __restrict__ Wx2,
    const float* __restrict__ Wx3, _Float16* __restrict__ wfrag,
    float* __restrict__ chis, float* __restrict__ pot_slot, int N)
{
    int tid = blockIdx.x * blockDim.x + threadIdx.x;
    if (tid < 3 * 64 * 64) {
        int e     = tid & 15;
        int lane  = (tid >> 4) & 31;
        int nb    = (tid >> 9) & 3;
        int kb    = (tid >> 11) & 1;
        int layer = tid >> 12;
        int k = kb * 32 + (lane >> 4) * 8 + (e & 7) + (e >> 3) * 16;
        int n = nb * 16 + (lane & 15);
        const float* W = (layer == 0) ? Wx1 : ((layer == 1) ? Wx2 : Wx3);
        wfrag[tid] = (_Float16)(W[k * 64 + n] * 0.125f);
    }
    for (int i = tid; i < N; i += gridDim.x * blockDim.x) chis[i] = 0.f;
    if (tid == 0) pot_slot[0] = 0.f;
}

// ---------------------------------------------------------------------------
// K1: per-edge chi MLP (48 ->16 silu -> 1) + atomic segment-sum to atoms.
// Weight indices are uniform -> scalar loads; per-thread 768 VALU FMAs.
// ---------------------------------------------------------------------------
__global__ void __launch_bounds__(256) k1_chi(
    const float* __restrict__ x, const int* __restrict__ senders,
    const float* __restrict__ Wc1, const float* __restrict__ Wc2,
    float* __restrict__ chis, int E)
{
    int e = blockIdx.x * blockDim.x + threadIdx.x;
    if (e >= E) return;
    const float s48 = 0.14433756729740643f;   // 1/sqrt(48)
    float acc[CE];
#pragma unroll
    for (int j = 0; j < CE; ++j) acc[j] = 0.f;
    const v4f* xr = (const v4f*)(x + (size_t)e * F_IN);
#pragma unroll
    for (int k4 = 0; k4 < F_IN / 4; ++k4) {
        v4f xv = xr[k4];
#pragma unroll
        for (int kk = 0; kk < 4; ++kk) {
            float xs = xv[kk] * s48;
            const int k = k4 * 4 + kk;
#pragma unroll
            for (int j = 0; j < CE; ++j)
                acc[j] = fmaf(xs, Wc1[k * CE + j], acc[j]);
        }
    }
    float chi = 0.f;
#pragma unroll
    for (int j = 0; j < CE; ++j)
        chi = fmaf(fast_silu(acc[j]), Wc2[j], chi);
    atomicAdd(&chis[senders[e]], chi * 0.25f); // 1/sqrt(16)
}

// ---------------------------------------------------------------------------
// K2: per-atom charge equilibration + charge-conditioned w row (17->16 linear)
//     + wave-reduced potential accumulation.
// ---------------------------------------------------------------------------
__global__ void __launch_bounds__(256) k2_atoms(
    const float* __restrict__ chis, const int* __restrict__ species,
    const float* __restrict__ radius, const float* __restrict__ hardness,
    const float* __restrict__ cemb, const float* __restrict__ chi_scale,
    const float* __restrict__ g_scale, const float* __restrict__ g_shift,
    const float* __restrict__ Ww, float* __restrict__ wrow,
    float* __restrict__ charges, float* __restrict__ pot, int N)
{
    int i = blockIdx.x * blockDim.x + threadIdx.x;
    float p = 0.f;
    if (i < N) {
        float chi = chis[i] * chi_scale[0];
        int sp = species[i];
        float gamma = fmaf(radius[sp], g_scale[0], g_shift[0]);
        float hx = hardness[sp];
        float hard = (hx > 20.f) ? hx : log1pf(__expf(hx));  // softplus
        float eta = hard + 1.f / gamma;
        float q = -chi / eta;
        charges[i] = q;
        p = chi * q + 0.5f * eta * q * q;
        const float s17 = 0.24253562503633297f;              // 1/sqrt(17)
        const float* emb = cemb + (size_t)sp * CE;
        float ev[CE];
#pragma unroll
        for (int c = 0; c < CE; ++c) ev[c] = emb[c];
#pragma unroll
        for (int j = 0; j < CE; ++j) {
            float a = q * Ww[j];
#pragma unroll
            for (int c = 0; c < CE; ++c)
                a = fmaf(ev[c], Ww[(1 + c) * CE + j], a);
            wrow[(size_t)i * CE + j] = a * s17;
        }
    }
#pragma unroll
    for (int off = 16; off > 0; off >>= 1) p += __shfl_down(p, off, 32);
    if ((threadIdx.x & 31) == 0) atomicAdd(pot, p);
}

// ---------------------------------------------------------------------------
// K3: one wave per 16-edge tile. 3x (16x64x64) WMMA GEMMs with silu, envelope
//     scale, plus coalesced passthrough copy of V. Weight fragments register-
//     resident; inter-layer activations restaged via padded LDS.
// ---------------------------------------------------------------------------
__global__ void __launch_bounds__(128) k3_edge_mlp(
    const float* __restrict__ x, const float* __restrict__ vecs,
    const float* __restrict__ V, const int* __restrict__ senders,
    const float* __restrict__ wrow, const _Float16* __restrict__ wfrag,
    float* __restrict__ xout, float* __restrict__ Vout, int E)
{
    __shared__ __align__(16) _Float16 hbuf[4][16 * HB_STRIDE];
    __shared__ float envb[4][16];

    const int lane = threadIdx.x & 31;
    const int wid  = threadIdx.x >> 5;
    const int m    = lane & 15;     // matrix row owned by this lane (A layout)
    const int hf   = lane >> 4;     // lane half-group
    const int wavesPerBlock = blockDim.x >> 5;
    const int gwave  = blockIdx.x * wavesPerBlock + wid;
    const int nwaves = gridDim.x * wavesPerBlock;

    // Load all 3 layers' B fragments (already fragment-ordered by k0_prep).
    v16h B[3][2][4];
#pragma unroll
    for (int L = 0; L < 3; ++L)
#pragma unroll
        for (int kb = 0; kb < 2; ++kb)
#pragma unroll
            for (int nb = 0; nb < 4; ++nb)
                B[L][kb][nb] = *(const v16h*)(wfrag + (((L * 2 + kb) * 4 + nb) * 32 + lane) * 16);

    _Float16* hb = hbuf[wid];
    float* eb = envb[wid];
    const v8f vzero = {0.f, 0.f, 0.f, 0.f, 0.f, 0.f, 0.f, 0.f};

    const int ntiles = (E + 15) >> 4;
    for (int t = gwave; t < ntiles; t += nwaves) {
        const int e0 = t << 4;
        int edge = e0 + m;
        if (edge >= E) edge = E - 1;

        // envelope(|r|/MAX_R), MAX_R = 1, by lanes 0..15
        if (hf == 0) {
            const float* vp = vecs + (size_t)edge * 3;
            float vx = vp[0], vy = vp[1], vz = vp[2];
            float u = sqrtf(vx * vx + vy * vy + vz * vz);
            float u2 = u * u, u6 = u2 * u2 * u2;
            eb[m] = (u < 1.f) ? (1.f - 28.f * u6 + 48.f * u6 * u - 21.f * u6 * u2) : 0.f;
        }

        // ---- layer-1 A fragments straight from global ----
        // kb=0: K 0..31  -> x[hf*8..+8) ++ x[16+hf*8..+8)
        // kb=1: K 32..63 -> x[32+hf*8..+8) ++ w[sender][hf*8..+8)
        const float* xr = x + (size_t)edge * F_IN;
        v4f c0  = *(const v4f*)(xr + hf * 8);
        v4f c0b = *(const v4f*)(xr + hf * 8 + 4);
        v4f c1  = *(const v4f*)(xr + 16 + hf * 8);
        v4f c1b = *(const v4f*)(xr + 16 + hf * 8 + 4);
        v4f c2  = *(const v4f*)(xr + 32 + hf * 8);
        v4f c2b = *(const v4f*)(xr + 32 + hf * 8 + 4);
        const int snd = senders[edge];
        const float* wp = wrow + (size_t)snd * CE + hf * 8;
        v4f c3  = *(const v4f*)(wp);
        v4f c3b = *(const v4f*)(wp + 4);

        v16h a0, a1;
#pragma unroll
        for (int j = 0; j < 4; ++j) {
            a0[j]      = (_Float16)c0[j];  a0[4 + j]  = (_Float16)c0b[j];
            a0[8 + j]  = (_Float16)c1[j];  a0[12 + j] = (_Float16)c1b[j];
            a1[j]      = (_Float16)c2[j];  a1[4 + j]  = (_Float16)c2b[j];
            a1[8 + j]  = (_Float16)c3[j];  a1[12 + j] = (_Float16)c3b[j];
        }

        // ---- layer 1 ----
        v8f acc[4];
#pragma unroll
        for (int nb = 0; nb < 4; ++nb) {
            v8f c = vzero;
            c = __builtin_amdgcn_wmma_f32_16x16x32_f16(false, a0, false, B[0][0][nb], (short)0, c, false, false);
            c = __builtin_amdgcn_wmma_f32_16x16x32_f16(false, a1, false, B[0][1][nb], (short)0, c, false, false);
            acc[nb] = c;
        }
#pragma unroll
        for (int nb = 0; nb < 4; ++nb)
#pragma unroll
            for (int v = 0; v < 8; ++v)
                hb[(v + 8 * hf) * HB_STRIDE + nb * 16 + m] = (_Float16)fast_silu(acc[nb][v]);
        asm volatile("s_wait_dscnt 0" ::: "memory");

        v16h a2[2];
#pragma unroll
        for (int kb = 0; kb < 2; ++kb) {
            v8h p0 = *(const v8h*)(hb + m * HB_STRIDE + kb * 32 + hf * 8);
            v8h p1 = *(const v8h*)(hb + m * HB_STRIDE + kb * 32 + 16 + hf * 8);
#pragma unroll
            for (int j = 0; j < 8; ++j) { a2[kb][j] = p0[j]; a2[kb][8 + j] = p1[j]; }
        }
        asm volatile("" ::: "memory");

        // ---- layer 2 ----
#pragma unroll
        for (int nb = 0; nb < 4; ++nb) {
            v8f c = vzero;
            c = __builtin_amdgcn_wmma_f32_16x16x32_f16(false, a2[0], false, B[1][0][nb], (short)0, c, false, false);
            c = __builtin_amdgcn_wmma_f32_16x16x32_f16(false, a2[1], false, B[1][1][nb], (short)0, c, false, false);
            acc[nb] = c;
        }
#pragma unroll
        for (int nb = 0; nb < 4; ++nb)
#pragma unroll
            for (int v = 0; v < 8; ++v)
                hb[(v + 8 * hf) * HB_STRIDE + nb * 16 + m] = (_Float16)fast_silu(acc[nb][v]);
        asm volatile("s_wait_dscnt 0" ::: "memory");

        v16h a3[2];
#pragma unroll
        for (int kb = 0; kb < 2; ++kb) {
            v8h p0 = *(const v8h*)(hb + m * HB_STRIDE + kb * 32 + hf * 8);
            v8h p1 = *(const v8h*)(hb + m * HB_STRIDE + kb * 32 + 16 + hf * 8);
#pragma unroll
            for (int j = 0; j < 8; ++j) { a3[kb][j] = p0[j]; a3[kb][8 + j] = p1[j]; }
        }
        asm volatile("" ::: "memory");

        // ---- layer 3 (no activation) ----
#pragma unroll
        for (int nb = 0; nb < 4; ++nb) {
            v8f c = vzero;
            c = __builtin_amdgcn_wmma_f32_16x16x32_f16(false, a3[0], false, B[2][0][nb], (short)0, c, false, false);
            c = __builtin_amdgcn_wmma_f32_16x16x32_f16(false, a3[1], false, B[2][1][nb], (short)0, c, false, false);
            acc[nb] = c;
        }

        float envv[8];
#pragma unroll
        for (int v = 0; v < 8; ++v) envv[v] = eb[v + 8 * hf];
#pragma unroll
        for (int nb = 0; nb < 4; ++nb)
#pragma unroll
            for (int v = 0; v < 8; ++v) {
                int row = v + 8 * hf;
                int ge = e0 + row;
                if (ge < E)
                    xout[(size_t)ge * H_DIM + nb * 16 + m] = acc[nb][v] * envv[v];
            }

        // ---- passthrough copy of V for this tile (coalesced b128) ----
        {
            const v4f* Vi = (const v4f*)V;
            v4f* Vo = (v4f*)Vout;
            size_t b4 = (size_t)e0 * 16;
            size_t lim = (size_t)E * 16;
#pragma unroll
            for (int i = 0; i < 8; ++i) {
                size_t idx = b4 + (size_t)i * 32 + lane;
                if (idx < lim) Vo[idx] = Vi[idx];
            }
        }
    }
}

// ---------------------------------------------------------------------------
extern "C" void kernel_launch(void* const* d_in, const int* in_sizes, int n_in,
                              void* d_out, int out_size, void* d_ws, size_t ws_size,
                              hipStream_t stream) {
    const float* vectors  = (const float*)d_in[0];
    const float* x        = (const float*)d_in[1];
    const float* V        = (const float*)d_in[2];
    const int*   senders  = (const int*)d_in[3];
    const int*   species  = (const int*)d_in[4];
    const float* radius   = (const float*)d_in[5];
    const float* hardness = (const float*)d_in[6];
    const float* cemb     = (const float*)d_in[7];
    const float* chi_s    = (const float*)d_in[8];
    const float* g_s      = (const float*)d_in[9];
    const float* g_sh     = (const float*)d_in[10];
    const float* Wc1      = (const float*)d_in[11];
    const float* Wc2      = (const float*)d_in[12];
    const float* Ww       = (const float*)d_in[13];
    const float* Wx1      = (const float*)d_in[14];
    const float* Wx2      = (const float*)d_in[15];
    const float* Wx3      = (const float*)d_in[16];

    const int E = in_sizes[1] / F_IN;
    const int N = in_sizes[4];

    float* xout    = (float*)d_out;
    float* Vout    = xout + (size_t)E * H_DIM;
    float* charges = Vout + (size_t)E * H_DIM;
    float* pot     = charges + N;

    float* chis      = (float*)d_ws;                       // [N]
    float* wrow      = chis + N;                           // [N*16]
    _Float16* wfrag  = (_Float16*)(wrow + (size_t)N * CE); // [3*64*64] f16

    int prepThreads = (3 * 64 * 64 > N) ? 3 * 64 * 64 : N;
    k0_prep<<<(prepThreads + 255) / 256, 256, 0, stream>>>(Wx1, Wx2, Wx3, wfrag, chis, pot, N);
    k1_chi<<<(E + 255) / 256, 256, 0, stream>>>(x, senders, Wc1, Wc2, chis, E);
    k2_atoms<<<(N + 255) / 256, 256, 0, stream>>>(chis, species, radius, hardness, cemb,
                                                  chi_s, g_s, g_sh, Ww, wrow, charges, pot, N);
    const int ntiles = (E + 15) >> 4;
    int blocks = (ntiles + 3) / 4;
    if (blocks > 2048) blocks = 2048;
    if (blocks < 1) blocks = 1;
    k3_edge_mlp<<<blocks, 128, 0, stream>>>(x, vectors, V, senders, wrow, wfrag, xout, Vout, E);
}
